// Decoder_83691732730277
// MI455X (gfx1250) — compile-verified, matrix operand order checked by reading, not run on hardware
//
#include <hip/hip_runtime.h>
#include <hip/hip_bf16.h>

// ---------------------------------------------------------------------------
// Transformer decoder (6 layers) for MI455X / gfx1250.
// Heavy GEMMs + attention run on v_wmma_f32_16x16x32_bf16 (wave32 WMMA),
// f32 accumulate; softmax / layernorm / residual stream kept in fp32.
// GEMM stages the block-shared weight tile in LDS via gfx1250 async
// global->LDS copies (ASYNCcnt double-buffered).
// ---------------------------------------------------------------------------

#define BATCH 8
#define SEQ   512
#define DMODEL 512
#define NHEAD 8
#define DHEAD 64
#define DFFN  2048
#define NLAYER 6
#define ROWS (BATCH * SEQ)          // 4096 token rows

typedef __bf16 bf16_t;
typedef __attribute__((ext_vector_type(16))) __bf16 v16bf;
typedef __attribute__((ext_vector_type(8)))  float  v8f;
typedef __attribute__((ext_vector_type(4)))  unsigned int u32x4;
typedef __attribute__((ext_vector_type(4)))  int v4i;

union Frag16 { v16bf v; unsigned int u[8]; };

#if defined(__AMDGCN__) && \
    __has_builtin(__builtin_amdgcn_global_load_async_to_lds_b128) && \
    __has_builtin(__builtin_amdgcn_s_wait_asynccnt)
#define HAVE_ASYNC_LDS 1
typedef __attribute__((address_space(1))) v4i glob_v4i;   // AS1: global
typedef __attribute__((address_space(3))) v4i lds_v4i;    // AS3: LDS
#else
#define HAVE_ASYNC_LDS 0
#endif

// K offset of bf16-pair j (j=0..7) inside a 16x32 (or 32x16) 16-bit fragment,
// per the CDNA5 ISA 16-bit A-matrix layout:
//   VGPR 0..3: K = {0..7}  + 8*half ;  VGPR 4..7: K = {16..23} + 8*half
__device__ __forceinline__ int kOff(int j, int hl) {
    return 2 * j + ((j >= 4) ? 8 : 0) + 8 * hl;
}

__device__ __forceinline__ unsigned int pack_bf16x2(float a, float b) {
    __bf16 x = (__bf16)a, y = (__bf16)b;
    unsigned short ux, uy;
    __builtin_memcpy(&ux, &x, 2);
    __builtin_memcpy(&uy, &y, 2);
    return (unsigned int)ux | ((unsigned int)uy << 16);
}

// ---------------------------------------------------------------------------
// Embedding + sinusoidal positional encoding -> x (f32) and xb (bf16)
// ---------------------------------------------------------------------------
__global__ void embed_kernel(const int* __restrict__ tok,
                             const float* __restrict__ emb,
                             float* __restrict__ x, bf16_t* __restrict__ xb) {
    int idx = blockIdx.x * blockDim.x + threadIdx.x;
    if (idx >= ROWS * DMODEL) return;
    int d  = idx & (DMODEL - 1);
    int rs = idx >> 9;              // b*SEQ + s
    int s  = rs & (SEQ - 1);
    int t  = tok[rs];
    // 10000^(-(d&~1)/512) = exp(-(d&~1) * ln(10000)/512)
    float ang = (float)s * expf(-(float)(d & ~1) * (9.210340371976184f / 512.0f));
    float pe  = (d & 1) ? cosf(ang) : sinf(ang);
    float v   = emb[(size_t)t * DMODEL + d] + pe;
    x[idx]  = v;
    xb[idx] = (bf16_t)v;
}

__global__ void cvt_kernel(const float* __restrict__ a, bf16_t* __restrict__ o, int n) {
    int idx = blockIdx.x * blockDim.x + threadIdx.x;
    if (idx < n) o[idx] = (bf16_t)a[idx];
}

// Pack one attention projection weight W[H, D, DK] (f32) -> Wt[N=512][K=512] bf16
__global__ void pack_qkv_kernel(const float* __restrict__ W, bf16_t* __restrict__ Wt) {
    int idx = blockIdx.x * blockDim.x + threadIdx.x;
    if (idx >= DMODEL * DMODEL) return;
    int k = idx & (DMODEL - 1);
    int n = idx >> 9;
    int h = n >> 6, c = n & 63;
    Wt[idx] = (bf16_t)W[((size_t)h * DMODEL + k) * DHEAD + c];
}

// Pack W[K, N] (f32, row-major) -> Wt[N][K] bf16 (transpose)
__global__ void pack_t_kernel(const float* __restrict__ W, bf16_t* __restrict__ Wt,
                              int K, int N) {
    int idx = blockIdx.x * blockDim.x + threadIdx.x;
    if (idx >= K * N) return;
    int k = idx % K;
    int n = idx / K;
    Wt[idx] = (bf16_t)W[(size_t)k * N + n];
}

// v [ROWS, 512] bf16 (heads concatenated) -> vT [B, H, DV, S] bf16
__global__ void vtrans_kernel(const bf16_t* __restrict__ vb, bf16_t* __restrict__ vT) {
    int idx = blockIdx.x * blockDim.x + threadIdx.x;
    if (idx >= ROWS * DMODEL) return;
    int s  = idx & (SEQ - 1);
    int r  = idx >> 9;             // (b*H+h)*64 + dv
    int dv = r & 63;
    int bh = r >> 6;
    int h = bh & 7, b = bh >> 3;
    vT[idx] = vb[((size_t)(b * SEQ + s)) * DMODEL + h * DHEAD + dv];
}

// ---------------------------------------------------------------------------
// WMMA GEMM:  C[M,N] = A[M,K](bf16, row-major) x Wt[N,K](bf16) + bias
// Block = 8 waves; block tile 128(M) x 64(N); K stepped by 32.
// The 64x32 B tile is shared by all 8 waves -> staged once per K-step in LDS
// (async global->LDS double-buffered when available). Row stride padded to
// 80 B so the 16-byte fragment reads are bank-conflict-free across lanes.
// M must be a multiple of 128 (all call sites satisfy this; no early return,
// keeping barriers valid and EXEC all-ones for WMMA).
// ---------------------------------------------------------------------------
#define BROW 80   // padded LDS row stride in bytes (row payload = 64 B)

__device__ __forceinline__ void gemm_tile(v8f acc[4], const bf16_t* ap, int k0,
                                          const char* brow, int lm, int hl) {
    Frag16 a;
    const bf16_t* p = ap + k0;
#pragma unroll
    for (int j = 0; j < 8; ++j)
        a.u[j] = *(const unsigned int*)(p + kOff(j, hl));
#pragma unroll
    for (int t = 0; t < 4; ++t) {
        Frag16 b;
        const char* rb = brow + (t * 16 + lm) * BROW + 16 * hl;
        ((u32x4*)b.u)[0] = *(const u32x4*)(rb);        // K = 8*hl .. 8*hl+7
        ((u32x4*)b.u)[1] = *(const u32x4*)(rb + 32);   // K = 16+8*hl .. 23+8*hl
        acc[t] = __builtin_amdgcn_wmma_f32_16x16x32_bf16(
            false, a.v, false, b.v, (short)0, acc[t], false, false);
    }
}

__global__ void __launch_bounds__(256)
gemm_wmma(const bf16_t* __restrict__ A,
          const bf16_t* __restrict__ Wt,
          const float* __restrict__ bias,
          float* __restrict__ outF,
          bf16_t* __restrict__ outB,
          int M, int N, int K, int relu) {
    __shared__ __align__(16) char Bt[2][64 * BROW];

    const int tid  = threadIdx.x;
    const int lane = tid & 31;
    const int wave = tid >> 5;
    const int hl   = lane >> 4;
    const int lm   = lane & 15;
    const int m0   = blockIdx.x * 128 + wave * 16;
    const int n0   = blockIdx.y * 64;

    // cooperative B-tile loader: 256 threads x 16 B = one 64x32 bf16 tile
    const int ln  = tid >> 2;                 // 0..63 : local n (row in tile)
    const int sub = tid & 3;                  // 16-byte chunk within the row
    const bf16_t* gsrc = Wt + (size_t)(n0 + ln) * K + sub * 8;
    char* ldst0 = &Bt[0][0] + ln * BROW + sub * 16;
    char* ldst1 = &Bt[1][0] + ln * BROW + sub * 16;

    v8f acc[4] = {};
    const bf16_t* ap = A + (size_t)(m0 + lm) * K;
    const int nt = K / 32;

#if HAVE_ASYNC_LDS
    __builtin_amdgcn_global_load_async_to_lds_b128(
        (glob_v4i*)(v4i*)gsrc, (lds_v4i*)(v4i*)ldst0, 0, 0);
    for (int i = 0; i < nt; ++i) {
        if (i + 1 < nt) {
            // overwrites the buffer last computed on at i-1; the barrier at the
            // end of iteration i-1 makes this safe.
            __builtin_amdgcn_global_load_async_to_lds_b128(
                (glob_v4i*)(v4i*)(gsrc + (i + 1) * 32),
                (lds_v4i*)(v4i*)(((i + 1) & 1) ? ldst1 : ldst0), 0, 0);
            __builtin_amdgcn_s_wait_asynccnt(1);   // tile i complete (in-order)
        } else {
            __builtin_amdgcn_s_wait_asynccnt(0);
        }
        __syncthreads();
        if (i + 2 < nt) __builtin_prefetch(ap + (i + 2) * 32, 0, 0);
        gemm_tile(acc, ap, i * 32, &Bt[i & 1][0], lm, hl);
        __syncthreads();
    }
#else
    for (int i = 0; i < nt; ++i) {
        __syncthreads();
        *(u32x4*)ldst0 = *(const u32x4*)(gsrc + i * 32);   // 16 B per thread
        __syncthreads();
        if (i + 2 < nt) __builtin_prefetch(ap + (i + 2) * 32, 0, 0);
        gemm_tile(acc, ap, i * 32, &Bt[0][0], lm, hl);
    }
#endif

#pragma unroll
    for (int t = 0; t < 4; ++t) {
        int n = n0 + t * 16 + lm;
        float bv = bias ? bias[n] : 0.0f;
#pragma unroll
        for (int r = 0; r < 8; ++r) {
            int m = m0 + r + 8 * hl;
            float val = acc[t][r] + bv;
            if (relu) val = fmaxf(val, 0.0f);
            if (outF) outF[(size_t)m * N + n] = val;
            if (outB) outB[(size_t)m * N + n] = (bf16_t)val;
        }
    }
}

// ---------------------------------------------------------------------------
// Fused attention: one wave per (b, h, 16-query tile).
// scores = (Q Kt)/8 + mask  -> LDS -> softmax (unnormalized exp, row sums)
// out    = (P Vt) * (1/rowsum)
// ---------------------------------------------------------------------------
#define SROW 516   // padded LDS row stride (floats) to avoid bank conflicts

__global__ void __launch_bounds__(32)
attn_wmma(const bf16_t* __restrict__ qb, const bf16_t* __restrict__ kb,
          const bf16_t* __restrict__ vT, const float* __restrict__ mask,
          float* __restrict__ out) {
    __shared__ float sc[16 * SROW];
    __shared__ float rowsum[16];

    const int lane = threadIdx.x & 31;
    const int hl   = lane >> 4;
    const int lm   = lane & 15;
    const int q0   = blockIdx.x * 16;
    const int h    = blockIdx.y;
    const int b    = blockIdx.z;

    // Q fragments (DK = 64 -> two K-chunks of 32)
    Frag16 qf[2];
    const bf16_t* qp = qb + ((size_t)(b * SEQ + q0 + lm)) * DMODEL + h * DHEAD;
#pragma unroll
    for (int c = 0; c < 2; ++c)
#pragma unroll
        for (int j = 0; j < 8; ++j)
            qf[c].u[j] = *(const unsigned int*)(qp + 32 * c + kOff(j, hl));

    const float scale = 0.125f;    // 1/sqrt(64)

    // ---- scores ----
    for (int jt = 0; jt < SEQ / 16; ++jt) {
        v8f acc = {};
#pragma unroll
        for (int c = 0; c < 2; ++c) {
            Frag16 kf;
            const bf16_t* kp = kb + ((size_t)(b * SEQ + jt * 16 + lm)) * DMODEL
                                  + h * DHEAD + 32 * c;
#pragma unroll
            for (int j = 0; j < 8; ++j)
                kf.u[j] = *(const unsigned int*)(kp + kOff(j, hl));
            acc = __builtin_amdgcn_wmma_f32_16x16x32_bf16(
                false, qf[c].v, false, kf.v, (short)0, acc, false, false);
        }
#pragma unroll
        for (int r = 0; r < 8; ++r) {
            int qrow = q0 + r + 8 * hl;
            int key  = jt * 16 + lm;
            float v = acc[r] * scale + mask[((size_t)b * SEQ + qrow) * SEQ + key];
            sc[(r + 8 * hl) * SROW + key] = v;
        }
    }
    __syncthreads();

    // ---- softmax (each lane: row = lm, columns [hl*256, hl*256+256)) ----
    {
        int row = lm;
        float mx = -3.0e38f;
        for (int c = hl * 256; c < hl * 256 + 256; ++c)
            mx = fmaxf(mx, sc[row * SROW + c]);
        mx = fmaxf(mx, __shfl_xor(mx, 16));
        float sum = 0.0f;
        for (int c = hl * 256; c < hl * 256 + 256; ++c) {
            float e = __expf(sc[row * SROW + c] - mx);
            sc[row * SROW + c] = e;
            sum += e;
        }
        sum += __shfl_xor(sum, 16);
        if (hl == 0) rowsum[row] = sum;
    }
    __syncthreads();

    // ---- P x V ----
    v8f acc[4] = {};
    for (int c = 0; c < SEQ / 32; ++c) {
        Frag16 a;
#pragma unroll
        for (int j = 0; j < 8; ++j) {
            int k = c * 32 + kOff(j, hl);
            a.u[j] = pack_bf16x2(sc[lm * SROW + k], sc[lm * SROW + k + 1]);
        }
#pragma unroll
        for (int t = 0; t < 4; ++t) {
            Frag16 bfr;
            const bf16_t* vp = vT + (((size_t)(b * NHEAD + h) * DHEAD + t * 16 + lm)) * SEQ
                                  + c * 32;
#pragma unroll
            for (int j = 0; j < 8; ++j)
                bfr.u[j] = *(const unsigned int*)(vp + kOff(j, hl));
            acc[t] = __builtin_amdgcn_wmma_f32_16x16x32_bf16(
                false, a.v, false, bfr.v, (short)0, acc[t], false, false);
        }
    }

#pragma unroll
    for (int t = 0; t < 4; ++t)
#pragma unroll
        for (int r = 0; r < 8; ++r) {
            int m = q0 + r + 8 * hl;
            float inv = 1.0f / rowsum[r + 8 * hl];
            out[((size_t)(b * SEQ + m)) * DMODEL + h * DHEAD + t * 16 + lm] =
                acc[t][r] * inv;
        }
}

// ---------------------------------------------------------------------------
// x = LN(x + y) * g + b ; also emits bf16 copy. One block (256 thr) per row.
// ---------------------------------------------------------------------------
__global__ void ln_kernel(float* __restrict__ x, const float* __restrict__ y,
                          const float* __restrict__ g, const float* __restrict__ bta,
                          bf16_t* __restrict__ xb) {
    __shared__ float s1[256], s2[256];
    int row = blockIdx.x;
    int t   = threadIdx.x;
    size_t base = (size_t)row * DMODEL;
    float v0 = x[base + t]       + y[base + t];
    float v1 = x[base + t + 256] + y[base + t + 256];
    s1[t] = v0 + v1;
    s2[t] = v0 * v0 + v1 * v1;
    __syncthreads();
    for (int s = 128; s > 0; s >>= 1) {
        if (t < s) { s1[t] += s1[t + s]; s2[t] += s2[t + s]; }
        __syncthreads();
    }
    float mean = s1[0] * (1.0f / DMODEL);
    float var  = s2[0] * (1.0f / DMODEL) - mean * mean;
    float rs   = rsqrtf(var + 1e-5f);
    float o0 = (v0 - mean) * rs * g[t]       + bta[t];
    float o1 = (v1 - mean) * rs * g[t + 256] + bta[t + 256];
    x[base + t]        = o0;
    x[base + t + 256]  = o1;
    xb[base + t]       = (bf16_t)o0;
    xb[base + t + 256] = (bf16_t)o1;
}

// ---------------------------------------------------------------------------
// Host orchestration
// ---------------------------------------------------------------------------
extern "C" void kernel_launch(void* const* d_in, const int* in_sizes, int n_in,
                              void* d_out, int out_size, void* d_ws, size_t ws_size,
                              hipStream_t stream) {
    (void)in_sizes; (void)n_in; (void)out_size; (void)ws_size;

    const int*   tok    = (const int*)  d_in[0];
    const float* enc    = (const float*)d_in[1];
    const float* smask  = (const float*)d_in[2];
    const float* cmask  = (const float*)d_in[3];
    const float* emb    = (const float*)d_in[4];
    const float* Wq_s   = (const float*)d_in[5];
    const float* Wk_s   = (const float*)d_in[6];
    const float* Wv_s   = (const float*)d_in[7];
    const float* bq_s   = (const float*)d_in[8];
    const float* bk_s   = (const float*)d_in[9];
    const float* bv_s   = (const float*)d_in[10];
    const float* Wq_c   = (const float*)d_in[11];
    const float* Wk_c   = (const float*)d_in[12];
    const float* Wv_c   = (const float*)d_in[13];
    const float* bq_c   = (const float*)d_in[14];
    const float* bk_c   = (const float*)d_in[15];
    const float* bv_c   = (const float*)d_in[16];
    const float* W1     = (const float*)d_in[17];
    const float* b1     = (const float*)d_in[18];
    const float* W2     = (const float*)d_in[19];
    const float* b2     = (const float*)d_in[20];
    const float* ln1_g  = (const float*)d_in[21];
    const float* ln1_b  = (const float*)d_in[22];
    const float* ln2_g  = (const float*)d_in[23];
    const float* ln2_b  = (const float*)d_in[24];

    // ---- workspace carve-up ----
    char* base = (char*)d_ws;
    size_t off = 0;
    auto alloc = [&](size_t bytes) -> char* {
        char* p = base + off;
        off += (bytes + 255) & ~(size_t)255;
        return p;
    };
    const size_t ACT_F = (size_t)ROWS * DMODEL * sizeof(float);   // 8 MB
    const size_t ACT_B = (size_t)ROWS * DMODEL * sizeof(bf16_t);  // 4 MB
    float*  x    = (float*) alloc(ACT_F);
    bf16_t* xb   = (bf16_t*)alloc(ACT_B);
    bf16_t* encb = (bf16_t*)alloc(ACT_B);
    bf16_t* qb   = (bf16_t*)alloc(ACT_B);
    bf16_t* kbuf = (bf16_t*)alloc(ACT_B);
    bf16_t* vbuf = (bf16_t*)alloc(ACT_B);
    bf16_t* vT   = (bf16_t*)alloc(ACT_B);
    float*  sa   = (float*) alloc(ACT_F);                          // attn/ffn out
    bf16_t* hid  = (bf16_t*)alloc((size_t)ROWS * DFFN * sizeof(bf16_t));
    const size_t WP = (size_t)DMODEL * DMODEL;                     // elems
    bf16_t* Wp  = (bf16_t*)alloc(NLAYER * 6 * WP * sizeof(bf16_t));
    bf16_t* W1t = (bf16_t*)alloc((size_t)NLAYER * DMODEL * DFFN * sizeof(bf16_t));
    bf16_t* W2t = (bf16_t*)alloc((size_t)NLAYER * DFFN * DMODEL * sizeof(bf16_t));

    // ---- weight packs (re-done each call: deterministic) ----
    const float* qkvSrc[6] = { Wq_s, Wk_s, Wv_s, Wq_c, Wk_c, Wv_c };
    for (int l = 0; l < NLAYER; ++l) {
        for (int p = 0; p < 6; ++p) {
            pack_qkv_kernel<<<(DMODEL * DMODEL) / 256, 256, 0, stream>>>(
                qkvSrc[p] + (size_t)l * NHEAD * DMODEL * DHEAD,
                Wp + ((size_t)l * 6 + p) * WP);
        }
        pack_t_kernel<<<(DMODEL * DFFN) / 256, 256, 0, stream>>>(
            W1 + (size_t)l * DMODEL * DFFN, W1t + (size_t)l * DMODEL * DFFN,
            DMODEL, DFFN);
        pack_t_kernel<<<(DFFN * DMODEL) / 256, 256, 0, stream>>>(
            W2 + (size_t)l * DFFN * DMODEL, W2t + (size_t)l * DFFN * DMODEL,
            DFFN, DMODEL);
    }
    cvt_kernel<<<(ROWS * DMODEL) / 256, 256, 0, stream>>>(enc, encb, ROWS * DMODEL);
    embed_kernel<<<(ROWS * DMODEL) / 256, 256, 0, stream>>>(tok, emb, x, xb);

    const dim3 gemmGrid512(ROWS / 128, DMODEL / 64);
    const dim3 gemmGridFFN(ROWS / 128, DFFN / 64);
    const dim3 attnGrid(SEQ / 16, NHEAD, BATCH);

    for (int l = 0; l < NLAYER; ++l) {
        bf16_t* Wl = Wp + (size_t)l * 6 * WP;

        // ---- self attention ----
        gemm_wmma<<<gemmGrid512, 256, 0, stream>>>(xb, Wl + 0 * WP, bq_s + l * DMODEL,
                                                   nullptr, qb, ROWS, DMODEL, DMODEL, 0);
        gemm_wmma<<<gemmGrid512, 256, 0, stream>>>(xb, Wl + 1 * WP, bk_s + l * DMODEL,
                                                   nullptr, kbuf, ROWS, DMODEL, DMODEL, 0);
        gemm_wmma<<<gemmGrid512, 256, 0, stream>>>(xb, Wl + 2 * WP, bv_s + l * DMODEL,
                                                   nullptr, vbuf, ROWS, DMODEL, DMODEL, 0);
        vtrans_kernel<<<(ROWS * DMODEL) / 256, 256, 0, stream>>>(vbuf, vT);
        attn_wmma<<<attnGrid, 32, 0, stream>>>(qb, kbuf, vT, smask, sa);
        ln_kernel<<<ROWS, 256, 0, stream>>>(x, sa, ln1_g + l * DMODEL,
                                            ln1_b + l * DMODEL, xb);

        // ---- cross attention (K/V from encoder output) ----
        gemm_wmma<<<gemmGrid512, 256, 0, stream>>>(xb,   Wl + 3 * WP, bq_c + l * DMODEL,
                                                   nullptr, qb, ROWS, DMODEL, DMODEL, 0);
        gemm_wmma<<<gemmGrid512, 256, 0, stream>>>(encb, Wl + 4 * WP, bk_c + l * DMODEL,
                                                   nullptr, kbuf, ROWS, DMODEL, DMODEL, 0);
        gemm_wmma<<<gemmGrid512, 256, 0, stream>>>(encb, Wl + 5 * WP, bv_c + l * DMODEL,
                                                   nullptr, vbuf, ROWS, DMODEL, DMODEL, 0);
        vtrans_kernel<<<(ROWS * DMODEL) / 256, 256, 0, stream>>>(vbuf, vT);
        attn_wmma<<<attnGrid, 32, 0, stream>>>(qb, kbuf, vT, cmask, sa);
        ln_kernel<<<ROWS, 256, 0, stream>>>(x, sa, ln2_g + l * DMODEL,
                                            ln2_b + l * DMODEL, xb);

        // ---- FFN ----
        gemm_wmma<<<gemmGridFFN, 256, 0, stream>>>(xb, W1t + (size_t)l * DMODEL * DFFN,
                                                   b1 + l * DFFN, nullptr, hid,
                                                   ROWS, DFFN, DMODEL, 1);
        gemm_wmma<<<gemmGrid512, 256, 0, stream>>>(hid, W2t + (size_t)l * DFFN * DMODEL,
                                                   b2 + l * DMODEL, sa, nullptr,
                                                   ROWS, DMODEL, DFFN, 0);
        // reference reuses ln2 for the third norm
        ln_kernel<<<ROWS, 256, 0, stream>>>(x, sa, ln2_g + l * DMODEL,
                                            ln2_b + l * DMODEL, xb);
    }

    (void)hipMemcpyAsync(d_out, x, ACT_F, hipMemcpyDeviceToDevice, stream);
}
